// EdgeDecoder_38233798869838
// MI455X (gfx1250) — matrix operand order repaired and checked by reading, not compiled, not run on hardware
//
#include <hip/hip_runtime.h>
#include <stdint.h>

// ---------------------------------------------------------------------------
// EdgeDecoder for MI455X (gfx1250, wave32, WMMA):
//   out[e] = relu(concat(zc[row[e]], za[col[e]]) @ W1 + b1) @ W2 + b2
// H = 128, 2H = 256.  bf16 WMMA (f32 accumulate), gather-bandwidth-bound.
//
// Roofline: ~131.6 GFLOP vs ~1.05 GB effective HBM traffic (z_article, 51MB,
// is L2-resident with ~20x reuse; z_customer streams with NT hint).
// 32 edges/block: each B tile feeds two WMMAs (halves W1 L2 refetch).
// ---------------------------------------------------------------------------

typedef __attribute__((ext_vector_type(16))) __bf16 v16bf;
typedef __attribute__((ext_vector_type(8)))  float  v8f;

struct U32x8 { uint4 lo; uint4 hi; };   // 32 bytes, bit-castable to v16bf

__device__ __forceinline__ unsigned short f32_to_bf16_rne(float f) {
  union { float f; uint32_t u; } c; c.f = f;
  uint32_t u = c.u;
  u += 0x7FFFu + ((u >> 16) & 1u);      // round-to-nearest-even
  return (unsigned short)(u >> 16);
}

__device__ __forceinline__ v16bf pack_v16bf(uint4 lo, uint4 hi) {
  U32x8 p; p.lo = lo; p.hi = hi;
  return __builtin_bit_cast(v16bf, p);
}

// ---------------------------------------------------------------------------
// Prep: swizzle W1 (f32 [256][128] row-major) into bf16 B-matrix layout.
// For N-tile t (0..7), K-step s (0..7), lane L (0..31), VGPR v (0..7):
//   N = t*16 + (L&15)
//   K = s*32 + (L>=16 ? 16 : 0) + 2*v + h      (h = packed bf16 half, 0/1)
// Stored as uint32 at Bsw[((t*8+s)*32 + L)*8 + v]  (64 KB total, L2-hot).
// ---------------------------------------------------------------------------
__global__ void edgedec_prep_w1(const float* __restrict__ W1,
                                uint32_t* __restrict__ Bsw) {
  int id = blockIdx.x * blockDim.x + threadIdx.x;   // 0 .. 16383
  int v = id & 7;
  int L = (id >> 3) & 31;
  int s = (id >> 8) & 7;
  int t = id >> 11;
  int K0 = s * 32 + ((L >= 16) ? 16 : 0) + 2 * v;
  int N  = t * 16 + (L & 15);
  uint32_t lo = f32_to_bf16_rne(W1[(size_t)K0 * 128 + N]);
  uint32_t hi = f32_to_bf16_rne(W1[(size_t)(K0 + 1) * 128 + N]);
  Bsw[id] = lo | (hi << 16);
}

// ---------------------------------------------------------------------------
// Main kernel: 32 edges per block, 256 threads = 8 wave32s.
// Wave w owns N-tile [w*16, w*16+16); computes TWO 16-edge M-tiles per
// K-step so each B fetch feeds two v_wmma_f32_16x16x32_bf16.
// ---------------------------------------------------------------------------
__global__ void __launch_bounds__(256)
edgedec_mlp(const float* __restrict__ zc, const float* __restrict__ za,
            const int* __restrict__ row, const int* __restrict__ col,
            const float* __restrict__ b1, const float* __restrict__ W2,
            const float* __restrict__ b2, const uint32_t* __restrict__ Bsw,
            float* __restrict__ out, int E) {
  __shared__ __align__(16) unsigned short sA[32 * 256];  // 32 edges x 256 K, bf16
  __shared__ int   sRow[32];
  __shared__ int   sCol[32];
  __shared__ float sPart[8][32];                         // wave x edge partials

  const int tid = threadIdx.x;
  const int e0  = blockIdx.x * 32;

  if (tid < 32) {
    int ge = e0 + tid; if (ge >= E) ge = E - 1;
    sRow[tid] = row[ge];
  } else if (tid < 64) {
    int j = tid - 32;
    int ge = e0 + j; if (ge >= E) ge = E - 1;
    sCol[j] = col[ge];
  }
  __syncthreads();

  // Coalesced gather + f32->bf16 convert into LDS (A staging, row-major
  // [edge][K] so each WMMA A operand is two ds_load_b128 per lane).
  // z_customer: streaming working set (512 MB) -> non-temporal loads so it
  // does not evict the L2-resident z_article table (51 MB, ~20x reuse).
  const int k = tid;                    // 0..255 = K index
  for (int j = 0; j < 32; ++j) {
    float val;
    if (k < 128) val = __builtin_nontemporal_load(&zc[(size_t)sRow[j] * 128 + k]);
    else         val = za[(size_t)sCol[j] * 128 + (k - 128)];
    sA[j * 256 + k] = f32_to_bf16_rne(val);
  }
  __syncthreads();

  const int w    = tid >> 5;            // wave id = N-tile
  const int lane = tid & 31;
  const int half = lane >> 4;           // 0: lanes 0-15, 1: lanes 16-31
  const int m    = lane & 15;           // A: M row group; B/C: N column

  v8f acc0 = {};                        // edges e0 + [0,16)
  v8f acc1 = {};                        // edges e0 + [16,32)
  const uint4* Bq = (const uint4*)Bsw;
  #pragma unroll
  for (int s = 0; s < 8; ++s) {
    // B (32x16 bf16): pre-swizzled, 32B contiguous per lane; feeds 2 WMMAs
    size_t bo = ((size_t)(w * 8 + s) * 32 + lane) * 2;    // in uint4 units
    uint4 b_lo = Bq[bo];
    uint4 b_hi = Bq[bo + 1];
    v16bf B = pack_v16bf(b_lo, b_hi);
    // A tiles (16x32 bf16 each): lane half selects the K+16 offset
    const int k0 = s * 32 + half * 16;
    uint4 a0_lo = *(const uint4*)&sA[m * 256 + k0];
    uint4 a0_hi = *(const uint4*)&sA[m * 256 + k0 + 8];
    uint4 a1_lo = *(const uint4*)&sA[(m + 16) * 256 + k0];
    uint4 a1_hi = *(const uint4*)&sA[(m + 16) * 256 + k0 + 8];
    v16bf A0 = pack_v16bf(a0_lo, a0_hi);
    v16bf A1 = pack_v16bf(a1_lo, a1_hi);
    acc0 = __builtin_amdgcn_wmma_f32_16x16x32_bf16(
        false, A0, false, B, (short)0, acc0, false, false);
    acc1 = __builtin_amdgcn_wmma_f32_16x16x32_bf16(
        false, A1, false, B, (short)0, acc1, false, false);
  }

  // Fused layer 2: bias + relu + W2 dot, reduced across the 16 N-lanes.
  const float b1n = b1[w * 16 + m];
  const float w2n = W2[w * 16 + m];
  #pragma unroll
  for (int v = 0; v < 8; ++v) {
    // C/D layout: M = v + half*8, N = m
    float h0 = fmaxf(acc0[v] + b1n, 0.0f);
    float h1 = fmaxf(acc1[v] + b1n, 0.0f);
    float p0 = h0 * w2n;
    float p1 = h1 * w2n;
    p0 += __shfl_xor(p0, 1, 32);  p1 += __shfl_xor(p1, 1, 32);
    p0 += __shfl_xor(p0, 2, 32);  p1 += __shfl_xor(p1, 2, 32);
    p0 += __shfl_xor(p0, 4, 32);  p1 += __shfl_xor(p1, 4, 32);
    p0 += __shfl_xor(p0, 8, 32);  p1 += __shfl_xor(p1, 8, 32);
    if (m == 0) {
      sPart[w][v + half * 8]      = p0;
      sPart[w][16 + v + half * 8] = p1;
    }
  }
  __syncthreads();

  if (tid < 32) {
    int ge = e0 + tid;
    if (ge < E) {
      float s = b2[0];
      #pragma unroll
      for (int ww = 0; ww < 8; ++ww) s += sPart[ww][tid];
      out[ge] = s;
    }
  }
}

// ---------------------------------------------------------------------------
extern "C" void kernel_launch(void* const* d_in, const int* in_sizes, int n_in,
                              void* d_out, int out_size, void* d_ws, size_t ws_size,
                              hipStream_t stream) {
  const float* zc  = (const float*)d_in[0];
  const float* za  = (const float*)d_in[1];
  const int*   row = (const int*)d_in[2];
  const int*   col = (const int*)d_in[3];
  const float* W1  = (const float*)d_in[4];
  const float* b1  = (const float*)d_in[5];
  const float* W2  = (const float*)d_in[6];
  const float* b2  = (const float*)d_in[7];
  float* out = (float*)d_out;
  const int E = in_sizes[2];

  uint32_t* Bsw = (uint32_t*)d_ws;      // 64 KB: swizzled bf16 W1

  edgedec_prep_w1<<<64, 256, 0, stream>>>(W1, Bsw);

  const int nblocks = (E + 31) / 32;
  edgedec_mlp<<<nblocks, 256, 0, stream>>>(zc, za, row, col, b1, W2, b2,
                                           Bsw, out, E);
}
